// DecoderBlock_2731599200794
// MI455X (gfx1250) — matrix operand order repaired and checked by reading, not compiled, not run on hardware
//
// MI455X (gfx1250, CDNA5) transformer decoder block.
// Compute-bound (~137 GFLOP vs ~3us of HBM traffic at 23.3 TB/s): all GEMMs +
// attention run on v_wmma_f32_16x16x32_bf16 (8x the K-depth of the only f32
// WMMA 16x16x4), f32 accumulation, f32 LN/softmax/residuals.
// GEMM tiles staged by the Tensor Data Mover (tensor_load_to_lds, double-
// buffered on TENSORcnt, A/B DMAs issued by different waves); attention is
// barrier-free flash attention: DPP row-max, base-2 exp, WMMA row sums.
#include <hip/hip_runtime.h>
#include <stddef.h>
#include <stdint.h>

#define B_  2
#define T_  2048
#define C_  1024
#define H_  16
#define DH_ 64
#define NT_ (B_*T_)                    // 4096 token rows
#define FF_ (4*C_)                     // 4096
#define QS_ ((size_t)B_*H_*T_*DH_)     // elems per q/k/v tensor

typedef __attribute__((ext_vector_type(16))) __bf16 v16bf;
typedef __attribute__((ext_vector_type(8)))  float  v8f;
typedef __attribute__((ext_vector_type(4)))  unsigned int v4u;
typedef __attribute__((ext_vector_type(8)))  unsigned int v8u;

union Frag16 { v16bf v; unsigned int u[8]; };

__device__ __forceinline__ v8f zero8() {
  v8f z = {0.f,0.f,0.f,0.f,0.f,0.f,0.f,0.f};
  return z;
}

// A-matrix fragment, 16x32 bf16, row-major source with leading dim `ld`.
// ISA: lane L holds row M=L%16; g=L/16: u[j] -> K=g*8+2j, u[4+j] -> K=16+g*8+2j.
__device__ __forceinline__ v16bf load_afrag(const __bf16* base, int ld) {
  const int lane = threadIdx.x & 31;
  const int r = lane & 15, g = lane >> 4;
  Frag16 f;
  const unsigned int* p0 = (const unsigned int*)(base + r*ld + g*8);
  const unsigned int* p1 = (const unsigned int*)(base + r*ld + 16 + g*8);
  #pragma unroll
  for (int j = 0; j < 4; ++j) { f.u[j] = p0[j]; f.u[4+j] = p1[j]; }
  return f.v;
}

// B-matrix fragment, 32x16 bf16, sourced from B^T (N x K row-major, ld).
// ISA: lane L holds col N=L%16; g=L/16: u[j] -> K=g*8+2j, u[4+j] -> K=16+g*8+2j.
__device__ __forceinline__ v16bf load_bfrag(const __bf16* baseT, int ld) {
  const int lane = threadIdx.x & 31;
  const int n = lane & 15, g = lane >> 4;
  Frag16 f;
  const unsigned int* p0 = (const unsigned int*)(baseT + n*ld + g*8);
  const unsigned int* p1 = (const unsigned int*)(baseT + n*ld + 16 + g*8);
  #pragma unroll
  for (int j = 0; j < 4; ++j) { f.u[j] = p0[j]; f.u[4+j] = p1[j]; }
  return f.v;
}

__device__ __forceinline__ v16bf ones_bfrag() {  // B matrix of all 1.0 (bf16)
  Frag16 f;
  #pragma unroll
  for (int j = 0; j < 8; ++j) f.u[j] = 0x3F803F80u;
  return f.v;
}

__device__ __forceinline__ v8f wmma_bf16(v16bf a, v16bf b, v8f c) {
  return __builtin_amdgcn_wmma_f32_16x16x32_bf16(false, a, false, b, (short)0, c,
                                                 false, false);
}

// max across each 16-lane half, pure VALU via DPP16 (no LDS, no dscnt waits):
// quad_perm(1,0,3,2)=xor1, quad_perm(2,3,0,1)=xor2, row_half_mirror, row_mirror.
__device__ __forceinline__ float dpp_max16(float x) {
  int t;
  t = __builtin_amdgcn_update_dpp(__float_as_int(x), __float_as_int(x),
                                  0x0B1, 0xF, 0xF, false);
  x = fmaxf(x, __int_as_float(t));
  t = __builtin_amdgcn_update_dpp(__float_as_int(x), __float_as_int(x),
                                  0x04E, 0xF, 0xF, false);
  x = fmaxf(x, __int_as_float(t));
  t = __builtin_amdgcn_update_dpp(__float_as_int(x), __float_as_int(x),
                                  0x141, 0xF, 0xF, false);
  x = fmaxf(x, __int_as_float(t));
  t = __builtin_amdgcn_update_dpp(__float_as_int(x), __float_as_int(x),
                                  0x140, 0xF, 0xF, false);
  x = fmaxf(x, __int_as_float(t));
  return x;
}

// ---- Tensor Data Mover: 2D bf16 tile (cols x rows) -> LDS --------------------
// D# per CDNA5 ISA: group0 {count=1, lds_addr, global_addr,
// type=2}; group1 {data_size=1(2B), tensor_dim0=tile_dim0=cols,
// tensor_dim1=tile_dim1=rows, tensor_dim0_stride=stride_elems}.
__device__ __forceinline__ void tdm_load_2d(const void* gaddr, unsigned lds_addr,
                                            unsigned cols, unsigned rows,
                                            unsigned stride_elems) {
  unsigned long long ga = (unsigned long long)(uintptr_t)gaddr;
  v4u g0;
  g0[0] = 1u;                                         // count=1 (user mode)
  g0[1] = lds_addr;                                   // LDS byte address
  g0[2] = (unsigned)ga;                               // global_addr[31:0]
  g0[3] = (unsigned)((ga >> 32) & 0x01FFFFFFull) | (2u << 30);  // addr hi | type=2
  v8u g1;
  g1[0] = 1u << 16;                                   // wg_mask=0, data_size=2B
  g1[1] = (cols & 0xFFFFu) << 16;                     // tensor_dim0 lo
  g1[2] = (cols >> 16) | ((rows & 0xFFFFu) << 16);    // dim0 hi | tensor_dim1 lo
  g1[3] = (rows >> 16) | ((cols & 0xFFFFu) << 16);    // dim1 hi | tile_dim0
  g1[4] = rows & 0xFFFFu;                             // tile_dim1 | tile_dim2=0
  g1[5] = stride_elems;                               // tensor_dim0_stride[31:0]
  g1[6] = 0u;
  g1[7] = 0u;
  asm volatile("tensor_load_to_lds %0, %1" :: "s"(g0), "s"(g1) : "memory");
}

__device__ __forceinline__ unsigned lds_byte_addr(const void* p) {
  return (unsigned)(uintptr_t)p;    // flat LDS aperture: addr[31:0] = LDS offset
}

// ---------------- weight conversion (one-shot) --------------------------------
__global__ void conv_qkv(const float* __restrict__ Wq, const float* __restrict__ Wk,
                         const float* __restrict__ Wv, __bf16* __restrict__ out) {
  size_t i = (size_t)blockIdx.x * 256 + threadIdx.x;   // 3*1024*1024 total
  int c  = (int)(i & 1023);
  int ng = (int)(i >> 10);
  int sel = ng >> 10;
  int nn  = ng & 1023;
  int h = nn >> 6, d = nn & 63;
  const float* W = (sel == 0) ? Wq : ((sel == 1) ? Wk : Wv);
  out[i] = (__bf16)W[((size_t)h*C_ + c)*DH_ + d];
}

__global__ void conv_t(const float* __restrict__ W, __bf16* __restrict__ out,
                       int K, int N) {
  size_t i = (size_t)blockIdx.x * 256 + threadIdx.x;   // N*K total
  size_t n = i / (size_t)K;
  size_t k = i % (size_t)K;
  out[i] = (__bf16)W[k*(size_t)N + n];
}

// ---------------- layer norm: f32 row -> bf16 row -----------------------------
__global__ void __launch_bounds__(256)
ln_kernel(const float* __restrict__ x, const float* __restrict__ gam,
          const float* __restrict__ bet, __bf16* __restrict__ out) {
  __shared__ float sm[8];
  const int row = blockIdx.x;
  const int tid = threadIdx.x;
  const float* xr = x + (size_t)row * C_;
  float4 xv = ((const float4*)xr)[tid];          // 256 threads * 4 = 1024
  float s = xv.x + xv.y + xv.z + xv.w;
  #pragma unroll
  for (int m = 16; m > 0; m >>= 1) s += __shfl_xor(s, m, 32);
  int w = tid >> 5, l = tid & 31;
  if (l == 0) sm[w] = s;
  __syncthreads();
  float tot = 0.f;
  #pragma unroll
  for (int i = 0; i < 8; ++i) tot += sm[i];
  float mu = tot * (1.f / C_);
  float dx = xv.x - mu, dy = xv.y - mu, dz = xv.z - mu, dw = xv.w - mu;
  float ss = dx*dx + dy*dy + dz*dz + dw*dw;
  #pragma unroll
  for (int m = 16; m > 0; m >>= 1) ss += __shfl_xor(ss, m, 32);
  __syncthreads();
  if (l == 0) sm[w] = ss;
  __syncthreads();
  float vtot = 0.f;
  #pragma unroll
  for (int i = 0; i < 8; ++i) vtot += sm[i];
  float rstd = rsqrtf(vtot * (1.f / C_) + 1e-5f);
  int c = tid * 4;
  size_t o = (size_t)row * C_ + c;
  out[o+0] = (__bf16)(dx * rstd * gam[c+0] + bet[c+0]);
  out[o+1] = (__bf16)(dy * rstd * gam[c+1] + bet[c+1]);
  out[o+2] = (__bf16)(dz * rstd * gam[c+2] + bet[c+2]);
  out[o+3] = (__bf16)(dw * rstd * gam[c+3] + bet[c+3]);
}

// ---------------- bf16 GEMM: C = A(MxK) * B(KxN), B given as B^T (NxK) --------
// 256 threads = 8 waves (2x4), WG tile 128x128, BK=64 (2 WMMA K-chunks/tile).
// TDM-staged, double-buffered on TENSORcnt: wave0 DMAs the A tile, wave1 the
// B tile for step k+1 while all waves run 16 WMMAs on step k.
// EPI 0: f32 out + bias + f32 residual       (proj / ffn2)
// EPI 1: bf16 out + bias + ReLU              (ffn1)
// EPI 2: bf16 QKV scatter; q,k -> (B,H,T,Dh), v -> (B,H,Dh,T) (pre-transposed)
template<int EPI>
__global__ void __launch_bounds__(256)
gemm_bf16(const __bf16* __restrict__ A, const __bf16* __restrict__ BT,
          int M, int N, int K,
          float* __restrict__ Cf, __bf16* __restrict__ Cb,
          const float* __restrict__ bias, const float* __restrict__ resid,
          __bf16* __restrict__ qkv) {
  __shared__ __align__(16) __bf16 As[2][128*64];   // 16KB per buffer
  __shared__ __align__(16) __bf16 Bs[2][128*64];
  const int n0 = blockIdx.x * 128;
  const int m0 = blockIdx.y * 128;
  const int wave = threadIdx.x >> 5, lane = threadIdx.x & 31;
  const int wm = wave >> 2, wn = wave & 3;     // 2 (M) x 4 (N) wave grid
  const int nl = lane & 15, g = lane >> 4;
  v8f acc[4][2];
  #pragma unroll
  for (int mt = 0; mt < 4; ++mt)
    #pragma unroll
    for (int nt = 0; nt < 2; ++nt) acc[mt][nt] = zero8();

  if (wave == 0)                                // prime tile 0 (A by wave0,
    tdm_load_2d(&A [(size_t)m0*K], lds_byte_addr(&As[0][0]), 64, 128, (unsigned)K);
  if (wave == 1)                                //  B by wave1: parallel DMAs)
    tdm_load_2d(&BT[(size_t)n0*K], lds_byte_addr(&Bs[0][0]), 64, 128, (unsigned)K);
  int cur = 0;
  for (int k0 = 0; k0 < K; k0 += 64) {
    const bool more = (k0 + 64 < K);
    if (more) {                                 // prefetch tile k+1 via TDM
      if (wave == 0)
        tdm_load_2d(&A [(size_t)m0*K + k0 + 64], lds_byte_addr(&As[cur^1][0]),
                    64, 128, (unsigned)K);
      if (wave == 1)
        tdm_load_2d(&BT[(size_t)n0*K + k0 + 64], lds_byte_addr(&Bs[cur^1][0]),
                    64, 128, (unsigned)K);
    }
    if (wave < 2) {                             // own tile resident (in-order TDM)
      if (more) __builtin_amdgcn_s_wait_tensorcnt(1);
      else      __builtin_amdgcn_s_wait_tensorcnt(0);
    }
    __syncthreads();                            // publish tile k to all waves
    #pragma unroll
    for (int c = 0; c < 2; ++c) {
      v16bf bf0 = load_bfrag(&Bs[cur][(wn*32     )*64 + c*32], 64);
      v16bf bf1 = load_bfrag(&Bs[cur][(wn*32 + 16)*64 + c*32], 64);
      #pragma unroll
      for (int mt = 0; mt < 4; ++mt) {
        v16bf af = load_afrag(&As[cur][(wm*64 + mt*16)*64 + c*32], 64);
        acc[mt][0] = wmma_bf16(af, bf0, acc[mt][0]);
        acc[mt][1] = wmma_bf16(af, bf1, acc[mt][1]);
      }
    }
    __syncthreads();                            // reads done before TDM reuses buf
    cur ^= 1;
  }
  // epilogue (C/D layout: lane holds col nl, rows r+8g)
  #pragma unroll
  for (int mt = 0; mt < 4; ++mt)
    #pragma unroll
    for (int nt = 0; nt < 2; ++nt)
      #pragma unroll
      for (int r = 0; r < 8; ++r) {
        int m = m0 + wm*64 + mt*16 + r + 8*g;
        int n = n0 + wn*32 + nt*16 + nl;
        float val = acc[mt][nt][r];
        if (EPI == 0) {
          val += bias[n] + resid[(size_t)m*N + n];
          Cf[(size_t)m*N + n] = val;
        } else if (EPI == 1) {
          val = fmaxf(val + bias[n], 0.f);
          Cb[(size_t)m*N + n] = (__bf16)val;
        } else {
          int sel = n >> 10, nn = n & 1023;
          int hh = nn >> 6, d = nn & 63;
          int bb = m >> 11, t = m & 2047;
          size_t idx;
          if (sel == 2)  // V stored transposed (B,H,Dh,T): native B^T for P*V
            idx = 2*QS_ + ((size_t)(bb*H_ + hh)*DH_ + d)*T_ + t;
          else
            idx = (size_t)sel*QS_ + ((size_t)(bb*H_ + hh)*T_ + t)*DH_ + d;
          qkv[idx] = (__bf16)val;
        }
      }
}

// ---------------- flash attention (barrier-free) ------------------------------
// Grid: B*H*(T/128); 8 waves/WG, each wave owns a 16-query tile; online softmax
// over 64-key blocks in base-2 domain: scale2 = C^-0.5 * log2(e).
// K (B,H,T,Dh) and Vt (B,H,Dh,T) are both native B^T sources -> B-fragments
// load straight from global (L2-resident: K+V per head = 512KB << 192MB L2).
// Row max: DPP16 butterfly (VALU only). Row sums: matrix engine (P x ones).
__global__ void __launch_bounds__(256)
attn_kernel(const __bf16* __restrict__ q, const __bf16* __restrict__ k,
            const __bf16* __restrict__ vt, __bf16* __restrict__ out) {
  __shared__ __align__(16) __bf16 pbuf[8][16*64];  // per-wave P relayout (C->A)
  const int qchunks = T_ / 128;
  const int qc = blockIdx.x % qchunks;
  const int bh = blockIdx.x / qchunks;
  const int b = bh / H_, h = bh % H_;
  const int wave = threadIdx.x >> 5;
  const int lane = threadIdx.x & 31;
  const int nl = lane & 15, g = lane >> 4;
  const int qbase = qc*128 + wave*16;
  const __bf16* qp  = q  + ((size_t)bh*T_ + qbase)*DH_;
  const __bf16* kb_ = k  + (size_t)bh*T_*DH_;
  const __bf16* vb_ = vt + (size_t)bh*DH_*T_;      // rows = d, ld = T
  v16bf qa0 = load_afrag(qp,      DH_);
  v16bf qa1 = load_afrag(qp + 32, DH_);
  const v16bf onesb = ones_bfrag();
  v8f o[4];
  float mi[8], li[8];
  #pragma unroll
  for (int i = 0; i < 4; ++i) o[i] = zero8();
  #pragma unroll
  for (int i = 0; i < 8; ++i) { mi[i] = -1e30f; li[i] = 0.f; }
  const float scale2 = 0.03125f * 1.4426950408889634f;  // C^-0.5 * log2(e)

  #pragma unroll 2
  for (int kb = 0; kb < T_; kb += 64) {
    // S = Q K^T (K row-major is already B^T layout)
    v8f s[4];
    #pragma unroll
    for (int nt = 0; nt < 4; ++nt) {
      const __bf16* kp = kb_ + (size_t)(kb + nt*16)*DH_;
      v8f a = zero8();
      a = wmma_bf16(qa0, load_bfrag(kp,      DH_), a);
      a = wmma_bf16(qa1, load_bfrag(kp + 32, DH_), a);
      s[nt] = a;
    }
    #pragma unroll
    for (int nt = 0; nt < 4; ++nt)
      #pragma unroll
      for (int r = 0; r < 8; ++r) s[nt][r] *= scale2;    // -> log2 domain
    // per-row max: per-lane max over tiles, then DPP butterfly (no LDS)
    float corr[8];
    #pragma unroll
    for (int r = 0; r < 8; ++r) {
      float rm = fmaxf(fmaxf(s[0][r], s[1][r]), fmaxf(s[2][r], s[3][r]));
      rm = dpp_max16(rm);
      float nm = fmaxf(mi[r], rm);
      corr[r] = __builtin_amdgcn_exp2f(mi[r] - nm);
      mi[r] = nm;
    }
    #pragma unroll
    for (int nt = 0; nt < 4; ++nt)
      #pragma unroll
      for (int r = 0; r < 8; ++r)
        s[nt][r] = __builtin_amdgcn_exp2f(s[nt][r] - mi[r]);
    #pragma unroll
    for (int dt = 0; dt < 4; ++dt)
      #pragma unroll
      for (int r = 0; r < 8; ++r) o[dt][r] *= corr[r];
    // P (C-layout) -> LDS -> A-fragment layout (same wave, no barrier)
    __bf16* pw = &pbuf[wave][0];
    #pragma unroll
    for (int nt = 0; nt < 4; ++nt)
      #pragma unroll
      for (int r = 0; r < 8; ++r)
        pw[(r + 8*g)*64 + nt*16 + nl] = (__bf16)s[nt][r];
    v16bf pa0 = load_afrag(pw,      64);
    v16bf pa1 = load_afrag(pw + 32, 64);
    // row sums on the matrix engine: every column of (P x ones) = row sum
    v8f rsum = zero8();
    rsum = wmma_bf16(pa0, onesb, rsum);
    rsum = wmma_bf16(pa1, onesb, rsum);
    #pragma unroll
    for (int r = 0; r < 8; ++r) li[r] = li[r]*corr[r] + rsum[r];
    // O += P * V   (B^T = Vt rows d, ld = T, k offset = kb)
    #pragma unroll
    for (int dt = 0; dt < 4; ++dt) {
      o[dt] = wmma_bf16(pa0, load_bfrag(vb_ + (size_t)(dt*16)*T_ + kb,      T_), o[dt]);
      o[dt] = wmma_bf16(pa1, load_bfrag(vb_ + (size_t)(dt*16)*T_ + kb + 32, T_), o[dt]);
    }
  }
  // normalize (one v_rcp per row) + store heads concatenated: (B,T,H*Dh) bf16
  float inv[8];
  #pragma unroll
  for (int r = 0; r < 8; ++r) inv[r] = __builtin_amdgcn_rcpf(li[r]);
  #pragma unroll
  for (int dt = 0; dt < 4; ++dt)
    #pragma unroll
    for (int r = 0; r < 8; ++r) {
      float val = o[dt][r] * inv[r];
      size_t m = (size_t)b*T_ + qbase + r + 8*g;
      out[m*(size_t)(H_*DH_) + h*DH_ + dt*16 + nl] = (__bf16)val;
    }
}

// ---------------- launch -------------------------------------------------------
extern "C" void kernel_launch(void* const* d_in, const int* in_sizes, int n_in,
                              void* d_out, int out_size, void* d_ws, size_t ws_size,
                              hipStream_t stream) {
  (void)in_sizes; (void)n_in; (void)out_size; (void)ws_size;
  const float* x   = (const float*)d_in[0];
  const float* Wq  = (const float*)d_in[1];
  const float* Wk  = (const float*)d_in[2];
  const float* Wv  = (const float*)d_in[3];
  const float* Wo  = (const float*)d_in[4];
  const float* bo  = (const float*)d_in[5];
  const float* W1  = (const float*)d_in[6];
  const float* b1  = (const float*)d_in[7];
  const float* W2  = (const float*)d_in[8];
  const float* b2  = (const float*)d_in[9];
  const float* g1  = (const float*)d_in[10];
  const float* be1 = (const float*)d_in[11];
  const float* g2  = (const float*)d_in[12];
  const float* be2 = (const float*)d_in[13];
  float* outp = (float*)d_out;

  char* ws = (char*)d_ws;
  size_t off = 0;
  auto alloc = [&](size_t bytes) -> char* {
    char* p = ws + off;
    off += (bytes + 255) & ~(size_t)255;
    return p;
  };
  __bf16* hbf    = (__bf16*)alloc((size_t)NT_*C_*2);        // ln1 out
  __bf16* h2bf   = (__bf16*)alloc((size_t)NT_*C_*2);        // ln2 out
  __bf16* WqkvT  = (__bf16*)alloc((size_t)3*C_*C_*2);       // (3*1024) x 1024
  __bf16* WoT    = (__bf16*)alloc((size_t)C_*C_*2);
  __bf16* W1T    = (__bf16*)alloc((size_t)FF_*C_*2);
  __bf16* W2T    = (__bf16*)alloc((size_t)C_*FF_*2);
  __bf16* qkvb   = (__bf16*)alloc((size_t)3*QS_*2);         // q,k,(v^T)
  __bf16* attnb  = (__bf16*)alloc((size_t)NT_*C_*2);        // concat heads
  float*  x1     = (float* )alloc((size_t)NT_*C_*4);        // residual after attn
  __bf16* actb   = (__bf16*)alloc((size_t)NT_*FF_*2);       // relu(h2 W1 + b1)

  // one-time bf16 weight repacks (B^T layout)
  conv_qkv<<<(3*C_*C_)/256, 256, 0, stream>>>(Wq, Wk, Wv, WqkvT);
  conv_t  <<<(C_*C_ )/256, 256, 0, stream>>>(Wo, WoT, C_,  C_ );
  conv_t  <<<(C_*FF_)/256, 256, 0, stream>>>(W1, W1T, C_,  FF_);
  conv_t  <<<(FF_*C_)/256, 256, 0, stream>>>(W2, W2T, FF_, C_ );

  // h = LN1(x)
  ln_kernel<<<NT_, 256, 0, stream>>>(x, g1, be1, hbf);
  // q,k,v = h @ Wqkv  (M=4096, N=3072, K=1024); v stored transposed
  gemm_bf16<2><<<dim3((3*C_)/128, NT_/128), 256, 0, stream>>>(
      hbf, WqkvT, NT_, 3*C_, C_, nullptr, nullptr, nullptr, nullptr, qkvb);
  // attention
  attn_kernel<<<B_*H_*(T_/128), 256, 0, stream>>>(qkvb, qkvb + QS_, qkvb + 2*QS_, attnb);
  // x1 = x + attn @ Wo + bo
  gemm_bf16<0><<<dim3(C_/128, NT_/128), 256, 0, stream>>>(
      attnb, WoT, NT_, C_, C_, x1, nullptr, bo, x, nullptr);
  // h2 = LN2(x1)
  ln_kernel<<<NT_, 256, 0, stream>>>(x1, g2, be2, h2bf);
  // a = relu(h2 @ W1 + b1)
  gemm_bf16<1><<<dim3(FF_/128, NT_/128), 256, 0, stream>>>(
      h2bf, W1T, NT_, FF_, C_, nullptr, actb, b1, nullptr, nullptr);
  // out = x1 + a @ W2 + b2
  gemm_bf16<0><<<dim3(C_/128, NT_/128), 256, 0, stream>>>(
      actb, W2T, NT_, C_, FF_, outp, nullptr, b2, x1, nullptr);
}
